// BiGRULM_76510547411093
// MI455X (gfx1250) — compile-verified, matrix operand order checked by reading, not run on hardware
//
#include <hip/hip_runtime.h>
#include <hip/hip_bf16.h>
#include <cstdint>
#include <math.h>

typedef __attribute__((ext_vector_type(16))) _Float16 v16h;
typedef __attribute__((ext_vector_type(8)))  float    v8f;

#define EMBD  512
#define RECD  128
#define SEQL  128
#define BATB  32
#define VOCAB 50257
#define G3    384          // 3*REC
#define NCHUNKS 393        // ceil(50257/128)
#define VPAD  50304        // 393*128

// ---------------------------------------------------------------------------
// Async global->LDS copy support (CDNA5 GLOBAL_LOAD_ASYNC_TO_LDS_*, ASYNCcnt)
// Param types per hipcc diagnostic: addrspace(1) ptr to int-vector (16B/8B).
// ---------------------------------------------------------------------------
#if defined(__has_builtin)
#  if __has_builtin(__builtin_amdgcn_global_load_async_to_lds_b128) && \
      __has_builtin(__builtin_amdgcn_global_load_async_to_lds_b64)
#    define HAVE_ASYNC_COPY 1
#  endif
#endif
#ifndef HAVE_ASYNC_COPY
#  define HAVE_ASYNC_COPY 0
#endif

#if HAVE_ASYNC_COPY
typedef int vs4i __attribute__((vector_size(16)));
typedef int vs2i __attribute__((vector_size(8)));
typedef __attribute__((address_space(1))) vs4i g4i;
typedef __attribute__((address_space(3))) vs4i l4i;
typedef __attribute__((address_space(1))) vs2i g2i;
typedef __attribute__((address_space(3))) vs2i l2i;
__device__ __forceinline__ void async_cp16(void* lds, const void* gmem) {
  __builtin_amdgcn_global_load_async_to_lds_b128((g4i*)gmem, (l4i*)lds, 0, 0);
}
__device__ __forceinline__ void async_cp8(void* lds, const void* gmem) {
  __builtin_amdgcn_global_load_async_to_lds_b64((g2i*)gmem, (l2i*)lds, 0, 0);
}
#  if __has_builtin(__builtin_amdgcn_s_wait_asynccnt)
#    define WAIT_ASYNC(n) __builtin_amdgcn_s_wait_asynccnt(n)
#  else
#    define WAIT_ASYNC(n) asm volatile("s_wait_asynccnt %0" :: "i"(n) : "memory")
#  endif
#endif

// ---------------------------------------------------------------------------
// WMMA fragment gathers (wave32, v_wmma_f32_16x16x32_f16)
// A 16x32 f16 layout (ISA 7.12.2): lane l<16 holds row M=l, halves map to
// K = {0..7,16..23}; lanes 16..31 hold same rows with K = {8..15,24..31}.
// B 32x16 mirrors A with N per lane. K pairs are contiguous -> b32/b128 reads.
// ---------------------------------------------------------------------------
__device__ __forceinline__ v16h frag_h_row32(const _Float16* row, int lane) {
  int kb = (lane & 16) ? 4 : 0;             // uint offset: 8 halves
  union { unsigned u[8]; v16h v; } r;
  const unsigned* p = (const unsigned*)row; // 32 halves = 16 uints
#pragma unroll
  for (int q = 0; q < 8; ++q) {
    int ku = (q & 3) + kb + ((q >> 2) * 8); // pair index
    r.u[q] = p[ku];
  }
  return r.v;
}

__device__ __forceinline__ v16h frag_f32_row32(const float* row, int lane) {
  int kb = (lane & 16) ? 8 : 0;
  union { _Float16 h[16]; v16h v; } r;
#pragma unroll
  for (int i = 0; i < 16; ++i) {
    int K = (i & 7) + kb + ((i >> 3) * 16);
    r.h[i] = (_Float16)row[K];
  }
  return r.v;
}

// ---------------------------------------------------------------------------
// K0: convert V (fp32, [50257,256]) -> f16 padded to [50304,256] (zero tail)
// ---------------------------------------------------------------------------
__global__ void k_cvt_v(const float* __restrict__ V, _Float16* __restrict__ Vh) {
  long total = (long)VPAD * 256;
  for (long i = (long)blockIdx.x * blockDim.x + threadIdx.x; i < total;
       i += (long)gridDim.x * blockDim.x) {
    long row = i >> 8;
    Vh[i] = (row < VOCAB) ? (_Float16)V[i] : (_Float16)0.f;
  }
}

// ---------------------------------------------------------------------------
// K1: gi[t,b,:] = emb[tok[t,b]] @ U^T + b1   (both directions)
// grid (256 M-tiles, 2 dirs), block 256 (8 waves, 3 N-tiles each of 24)
// ---------------------------------------------------------------------------
__global__ void k_gi(const int* __restrict__ tok, const float* __restrict__ Emb,
                     const float* __restrict__ U,  const float* __restrict__ Ub,
                     const float* __restrict__ b1, const float* __restrict__ b1b,
                     float* __restrict__ GIf, float* __restrict__ GIb) {
  __shared__ _Float16 Ah[16 * 32];
  __shared__ _Float16 Uh[G3 * 32];
  __shared__ int toks[16];
  const int mt = blockIdx.x, dir = blockIdx.y;
  const float* Us = dir ? Ub : U;
  const float* bs = dir ? b1b : b1;
  float* GO = dir ? GIb : GIf;
  const int tid = threadIdx.x, lane = tid & 31, w = tid >> 5;
  if (tid < 16) toks[tid] = tok[mt * 16 + tid];
  __syncthreads();
  v8f acc0 = {}, acc1 = {}, acc2 = {};
  const int nt0 = w * 3, nt1 = w * 3 + 1, nt2 = w * 3 + 2;
  for (int kc = 0; kc < 16; ++kc) {
    for (int e = tid; e < 16 * 32; e += 256) {
      int r = e >> 5, k = e & 31;
      Ah[e] = (_Float16)Emb[(long)toks[r] * EMBD + kc * 32 + k];
    }
    for (int e = tid; e < G3 * 32; e += 256) {
      int n = e >> 5, k = e & 31;
      Uh[e] = (_Float16)Us[n * EMBD + kc * 32 + k];
    }
    __syncthreads();
    v16h a  = frag_h_row32(&Ah[(lane & 15) * 32], lane);
    v16h b0 = frag_h_row32(&Uh[(nt0 * 16 + (lane & 15)) * 32], lane);
    v16h b1f= frag_h_row32(&Uh[(nt1 * 16 + (lane & 15)) * 32], lane);
    v16h b2f= frag_h_row32(&Uh[(nt2 * 16 + (lane & 15)) * 32], lane);
    acc0 = __builtin_amdgcn_wmma_f32_16x16x32_f16(false, a, false, b0, (short)0, acc0, false, false);
    acc1 = __builtin_amdgcn_wmma_f32_16x16x32_f16(false, a, false, b1f,(short)0, acc1, false, false);
    acc2 = __builtin_amdgcn_wmma_f32_16x16x32_f16(false, a, false, b2f,(short)0, acc2, false, false);
    __syncthreads();
  }
  const int colL = lane & 15, rb = 8 * (lane >> 4);
  const int c0 = nt0 * 16 + colL, c1 = nt1 * 16 + colL, c2 = nt2 * 16 + colL;
  const float bb0 = bs[c0], bb1 = bs[c1], bb2 = bs[c2];
#pragma unroll
  for (int v = 0; v < 8; ++v) {
    long m = (long)(mt * 16 + rb + v) * G3;
    GO[m + c0] = acc0[v] + bb0;
    GO[m + c1] = acc1[v] + bb1;
    GO[m + c2] = acc2[v] + bb2;
  }
}

// ---------------------------------------------------------------------------
// K2: sequential GRU recurrence, one workgroup per direction.
// W resident in LDS as f16; h resident in LDS; gh = h@W^T via WMMA.
// Writes total_h (f16, [L,B,256]) including the zero boundary rows.
// ---------------------------------------------------------------------------
#define RNN_LDS (G3*RECD*2 + (BATB*G3 + BATB*RECD + G3)*4)
__global__ void __launch_bounds__(768, 1) k_rnn(
    const float* __restrict__ GIf, const float* __restrict__ GIb,
    const float* __restrict__ W,   const float* __restrict__ Wb,
    const float* __restrict__ b2,  const float* __restrict__ b2b,
    _Float16* __restrict__ TH) {
  extern __shared__ char smem[];
  _Float16* Wh = (_Float16*)smem;                 // 384*128 f16
  float* ghs = (float*)(smem + G3 * RECD * 2);    // 32*384
  float* hf  = ghs + BATB * G3;                   // 32*128
  float* b2s = hf + BATB * RECD;                  // 384
  const int dir = blockIdx.x;
  const float* GI = dir ? GIb : GIf;
  const float* Ws = dir ? Wb : W;
  const float* bb = dir ? b2b : b2;
  const int tid = threadIdx.x, lane = tid & 31, w = tid >> 5; // 24 waves
  const int coff = dir ? RECD : 0;
  for (int e = tid; e < G3 * RECD; e += 768) Wh[e] = (_Float16)Ws[e];
  for (int e = tid; e < BATB * RECD; e += 768) hf[e] = 0.f;
  if (tid < G3) b2s[tid] = bb[tid];
  { // zero boundary rows (fwd t=0 cols 0..127, bwd t=L-1 cols 128..255)
    int tz = dir ? (SEQL - 1) : 0;
    for (int e = tid; e < BATB * RECD; e += 768) {
      int b = e >> 7, j = e & 127;
      TH[(tz * BATB + b) * 2 * RECD + coff + j] = (_Float16)0.f;
    }
  }
  __syncthreads();
  for (int s = 0; s < SEQL - 1; ++s) {
    const int tsrc = dir ? (SEQL - 1 - s) : s;
    const int tdst = dir ? (SEQL - 2 - s) : (s + 1);
    const float* gi = GI + (long)tsrc * BATB * G3;
    if (s + 1 < SEQL - 1) {    // prefetch next step's gi (49KB) into caches
      int tn = dir ? (SEQL - 2 - s) : (s + 1);
      const char* p = (const char*)(GI + (long)tn * BATB * G3);
      __builtin_prefetch(p + tid * 64, 0, 0);
    }
    // gh = h @ W^T : 48 tiles (2 M x 24 N), 2 per wave, K = 4 chunks of 32
#pragma unroll
    for (int tt = 0; tt < 2; ++tt) {
      const int mtb = tt, nt = w;
      v8f acc = {};
#pragma unroll
      for (int kc = 0; kc < 4; ++kc) {
        v16h a = frag_f32_row32(&hf[(mtb * 16 + (lane & 15)) * RECD + kc * 32], lane);
        v16h b = frag_h_row32(&Wh[(nt * 16 + (lane & 15)) * RECD + kc * 32], lane);
        acc = __builtin_amdgcn_wmma_f32_16x16x32_f16(false, a, false, b, (short)0, acc, false, false);
      }
      const int colL = lane & 15, rb = 8 * (lane >> 4);
#pragma unroll
      for (int v = 0; v < 8; ++v)
        ghs[(mtb * 16 + rb + v) * G3 + nt * 16 + colL] = acc[v];
    }
    __syncthreads();
    // gates: z=sig(i_i+h_i) r=sig(i_r+h_r) n=tanh(i_n+r*h_n) h'=n+z*(h-n)
    for (int e = tid; e < BATB * RECD; e += 768) {
      const int b = e >> 7, j = e & 127;
      const float gir = gi[b * G3 + j];
      const float gii = gi[b * G3 + RECD + j];
      const float gin = gi[b * G3 + 2 * RECD + j];
      const float ghr = ghs[b * G3 + j] + b2s[j];
      const float ghi = ghs[b * G3 + RECD + j] + b2s[RECD + j];
      const float ghn = ghs[b * G3 + 2 * RECD + j] + b2s[2 * RECD + j];
      const float z = 1.f / (1.f + expf(-(gii + ghi)));
      const float r = 1.f / (1.f + expf(-(gir + ghr)));
      const float hp = hf[e];
      const float n = tanhf(gin + r * ghn);
      const float hn = n + z * (hp - n);
      hf[e] = hn;
      TH[(tdst * BATB + b) * 2 * RECD + coff + j] = (_Float16)hn;
    }
    __syncthreads();
  }
}

// ---------------------------------------------------------------------------
// K3: logits tile GEMM (M=16 rows x N=128 cols, K=256) + raw logits to d_out
//     + per-block (rowmax, sumexp) partials. V f16 serves from L2 (26MB<192MB).
//     K staged in 4 slabs of 64; async global->LDS DMA double-buffered.
// ---------------------------------------------------------------------------
__global__ void k_logits(const _Float16* __restrict__ TH,
                         const _Float16* __restrict__ Vh,
                         float* __restrict__ out, float* __restrict__ part) {
  __shared__ _Float16 Ah[2][16 * 64];
  __shared__ _Float16 Bh[2][128 * 64];
  __shared__ float tile[16 * 128];
  __shared__ float red[16 * 17];
  __shared__ float rmax[16];
  const int mt = blockIdx.x, nc = blockIdx.y;
  const int tid = threadIdx.x, lane = tid & 31, w = tid >> 5; // 8 waves
  const _Float16* Abase = TH + (long)mt * 16 * 256;
  const _Float16* Bbase = Vh + (long)nc * 128 * 256;
  v8f acc = {};
#if HAVE_ASYNC_COPY
  // Each thread issues exactly 5 async ops per K-stage:
  //   1x b64  : A tile 16x64 halves  (256 x 8B)
  //   4x b128 : B tile 128x64 halves (1024 x 16B)
  auto issue = [&](int ks) {
    const int buf = ks & 1;
    {
      int r = tid >> 4, k = tid & 15;
      async_cp8(&Ah[buf][r * 64 + k * 4], Abase + r * 256 + ks * 64 + k * 4);
    }
#pragma unroll
    for (int it = 0; it < 4; ++it) {
      int e = tid + it * 256;
      int r = e >> 3, k = e & 7;
      async_cp16(&Bh[buf][r * 64 + k * 8], Bbase + r * 256 + ks * 64 + k * 8);
    }
  };
  issue(0);
  for (int ks = 0; ks < 4; ++ks) {
    if (ks < 3) { issue(ks + 1); WAIT_ASYNC(5); }
    else        { WAIT_ASYNC(0); }
    __syncthreads();
    const int buf = ks & 1;
#pragma unroll
    for (int kc = 0; kc < 2; ++kc) {
      v16h a = frag_h_row32(&Ah[buf][(lane & 15) * 64 + kc * 32], lane);
      v16h b = frag_h_row32(&Bh[buf][(w * 16 + (lane & 15)) * 64 + kc * 32], lane);
      acc = __builtin_amdgcn_wmma_f32_16x16x32_f16(false, a, false, b, (short)0, acc, false, false);
    }
    __syncthreads();
  }
#else
  for (int ks = 0; ks < 4; ++ks) {
    for (int e = tid; e < 16 * 8; e += 256) {
      int r = e >> 3, k = e & 7;
      ((uint4*)Ah[0])[e] = *(const uint4*)(Abase + r * 256 + ks * 64 + k * 8);
    }
    for (int e = tid; e < 128 * 8; e += 256) {
      int r = e >> 3, k = e & 7;
      ((uint4*)Bh[0])[e] = *(const uint4*)(Bbase + r * 256 + ks * 64 + k * 8);
    }
    __syncthreads();
#pragma unroll
    for (int kc = 0; kc < 2; ++kc) {
      v16h a = frag_h_row32(&Ah[0][(lane & 15) * 64 + kc * 32], lane);
      v16h b = frag_h_row32(&Bh[0][(w * 16 + (lane & 15)) * 64 + kc * 32], lane);
      acc = __builtin_amdgcn_wmma_f32_16x16x32_f16(false, a, false, b, (short)0, acc, false, false);
    }
    __syncthreads();
  }
#endif
  const int colL = lane & 15, rb = 8 * (lane >> 4);
  const int colg = nc * 128 + w * 16 + colL;
#pragma unroll
  for (int v = 0; v < 8; ++v) {
    const int m = rb + v;
    tile[m * 128 + w * 16 + colL] = acc[v];
    if (colg < VOCAB)
      out[(long)(mt * 16 + m) * VOCAB + colg] = acc[v];
  }
  __syncthreads();
  // per-row stats over this block's (<=128) columns
  int Nv = VOCAB - nc * 128; if (Nv > 128) Nv = 128;
  const int r = tid & 15, seg = tid >> 4;    // 16 segs of 8 cols
  float mx = -__builtin_inff();
  for (int c = seg * 8; c < seg * 8 + 8 && c < Nv; ++c)
    mx = fmaxf(mx, tile[r * 128 + c]);
  red[r * 17 + seg] = mx;
  __syncthreads();
  if (tid < 16) {
    float m2 = red[tid * 17];
    for (int s2 = 1; s2 < 16; ++s2) m2 = fmaxf(m2, red[tid * 17 + s2]);
    rmax[tid] = m2;
  }
  __syncthreads();
  const float M = rmax[r];
  float sm = 0.f;
  for (int c = seg * 8; c < seg * 8 + 8 && c < Nv; ++c)
    sm += expf(tile[r * 128 + c] - M);
  red[r * 17 + seg] = sm;
  __syncthreads();
  if (tid < 16) {
    float s2 = 0.f;
    for (int sgg = 0; sgg < 16; ++sgg) s2 += red[tid * 17 + sgg];
    float* pp = part + ((long)(mt * 16 + tid) * NCHUNKS + nc) * 2;
    pp[0] = rmax[tid]; pp[1] = s2;
  }
}

// ---------------------------------------------------------------------------
// K4: merge per-chunk (max, sumexp) -> logZ[row] = M + log(sum s_j e^{m_j-M})
// ---------------------------------------------------------------------------
__global__ void k_merge(const float* __restrict__ part, float* __restrict__ logZ) {
  __shared__ float sm[128];
  const int row = blockIdx.x, tid = threadIdx.x;
  const float* pr = part + (long)row * NCHUNKS * 2;
  float mx = -__builtin_inff();
  for (int c = tid; c < NCHUNKS; c += 128) mx = fmaxf(mx, pr[c * 2]);
  sm[tid] = mx; __syncthreads();
  for (int st = 64; st > 0; st >>= 1) {
    if (tid < st) sm[tid] = fmaxf(sm[tid], sm[tid + st]);
    __syncthreads();
  }
  const float M = sm[0]; __syncthreads();
  float s = 0.f;
  for (int c = tid; c < NCHUNKS; c += 128) s += pr[c * 2 + 1] * expf(pr[c * 2] - M);
  sm[tid] = s; __syncthreads();
  for (int st = 64; st > 0; st >>= 1) {
    if (tid < st) sm[tid] += sm[tid + st];
    __syncthreads();
  }
  if (tid == 0) logZ[row] = M + logf(sm[0]);
}

// ---------------------------------------------------------------------------
// K5: out[row, c] -= logZ[row]
// ---------------------------------------------------------------------------
__global__ void k_norm(float* __restrict__ out, const float* __restrict__ logZ) {
  const int row = blockIdx.x;
  const float lz = logZ[row];
  const long base = (long)row * VOCAB;
  int c = blockIdx.y * 1024 + threadIdx.x;
#pragma unroll
  for (int it = 0; it < 4; ++it, c += 256)
    if (c < VOCAB) out[base + c] -= lz;
}

// ---------------------------------------------------------------------------
extern "C" void kernel_launch(void* const* d_in, const int* in_sizes, int n_in,
                              void* d_out, int out_size, void* d_ws, size_t ws_size,
                              hipStream_t stream) {
  const int*   tok = (const int*)  d_in[0];
  const float* Emb = (const float*)d_in[1];
  const float* U   = (const float*)d_in[2];
  const float* W   = (const float*)d_in[3];
  const float* Ub  = (const float*)d_in[4];
  const float* Wb  = (const float*)d_in[5];
  const float* V   = (const float*)d_in[6];
  const float* b1  = (const float*)d_in[7];
  const float* b2  = (const float*)d_in[8];
  const float* b1b = (const float*)d_in[9];
  const float* b2b = (const float*)d_in[10];
  float* out = (float*)d_out;

  // workspace carve-up (256B aligned)
  char* ws = (char*)d_ws;
  size_t o = 0;
  auto carve = [&](size_t bytes) { size_t r = o; o += (bytes + 255) & ~(size_t)255; return r; };
  _Float16* Vh  = (_Float16*)(ws + carve((size_t)VPAD * 256 * 2));      // 25.8MB
  float*    GIf = (float*)   (ws + carve((size_t)SEQL * BATB * G3 * 4));// 6.3MB
  float*    GIb = (float*)   (ws + carve((size_t)SEQL * BATB * G3 * 4));// 6.3MB
  _Float16* TH  = (_Float16*)(ws + carve((size_t)SEQL * BATB * 256 * 2));// 2MB
  float*    part= (float*)   (ws + carve((size_t)4096 * NCHUNKS * 2 * 4));// 12.9MB
  float*    logZ= (float*)   (ws + carve((size_t)4096 * 4));

  k_cvt_v<<<2048, 256, 0, stream>>>(V, Vh);
  k_gi<<<dim3(256, 2), 256, 0, stream>>>(tok, Emb, U, Ub, b1, b1b, GIf, GIb);
  (void)hipFuncSetAttribute((const void*)k_rnn,
      hipFuncAttributeMaxDynamicSharedMemorySize, RNN_LDS);
  k_rnn<<<2, 768, RNN_LDS, stream>>>(GIf, GIb, W, Wb, b2, b2b, TH);
  k_logits<<<dim3(256, NCHUNKS), 256, 0, stream>>>(TH, Vh, out, part);
  k_merge<<<4096, 128, 0, stream>>>(part, logZ);
  k_norm<<<dim3(4096, 50), 256, 0, stream>>>(out, logZ);
}
